// BaseLinearSSM_87823491268725
// MI455X (gfx1250) — matrix operand order repaired and checked by reading, not compiled, not run on hardware
//
#include <hip/hip_runtime.h>

typedef __attribute__((ext_vector_type(16))) __bf16 v16bf;
typedef __attribute__((ext_vector_type(8)))  __bf16 v8bf;
typedef __attribute__((ext_vector_type(8)))  float  v8f;
typedef __attribute__((ext_vector_type(4)))  unsigned int u32x4;
typedef __attribute__((ext_vector_type(8)))  int   i32x8;
typedef __attribute__((ext_vector_type(4)))  int   i32x4;

#define T_LEN   2048
#define BATCHSZ 16
#define IN_F    128
#define OUT_F   128
#define NSTATE  512
#define SROWS   1024          /* 2*NSTATE stacked (re;im) */
#define NWG     16
#define RPW     64            /* rows per workgroup: 1024/16 */

#if __has_builtin(__builtin_amdgcn_s_cluster_barrier) && __has_attribute(amdgpu_cluster_dims)
#define USE_CLUSTER 1
#define CLUSTER_ATTR __attribute__((amdgpu_cluster_dims(NWG, 1, 1)))
#else
#define USE_CLUSTER 0
#define CLUSTER_ATTR
#endif

#if __has_builtin(__builtin_amdgcn_tensor_load_to_lds)
#define HAVE_TDM 1
#else
#define HAVE_TDM 0
#endif

// ---------- operand builders (CDNA5 WMMA VGPR layouts, wave32) ----------

// A-operand: element j <- k = base + 16*h + j; two 16B runs.
__device__ __forceinline__ v16bf a_from_lds(const __bf16* rowp /* row + kt*32 + 16*h */) {
  v8bf lo = *(const v8bf*)(rowp);
  v8bf hi = *(const v8bf*)(rowp + 8);
  v16bf r;
#pragma unroll
  for (int j = 0; j < 8; ++j) { r[j] = lo[j]; r[j + 8] = hi[j]; }
  return r;
}

__device__ __forceinline__ v16bf a_from_f32(const float* __restrict__ rowp, int c0) {
  v16bf r;
#pragma unroll
  for (int j = 0; j < 16; ++j) {
    int k = c0 + j;
    r[j] = (__bf16)rowp[k];
  }
  return r;
}

// A-operand for Cmat = [C_re | -C_im]
__device__ __forceinline__ v16bf a_from_cmat(const float* __restrict__ C_re,
                                             const float* __restrict__ C_im,
                                             int o, int c0) {
  v16bf r;
#pragma unroll
  for (int half = 0; half < 2; ++half) {
    int k0 = c0 + half * 8;
    const float* p; float s;
    if (k0 < NSTATE) { p = C_re + (size_t)o * NSTATE + k0;            s =  1.f; }
    else             { p = C_im + (size_t)o * NSTATE + (k0 - NSTATE); s = -1.f; }
#pragma unroll
    for (int j = 0; j < 8; ++j) r[half * 8 + j] = (__bf16)(s * p[j]);
  }
  return r;
}

// B-operand: element j <- k = base + 16*h + j (16 contiguous)
__device__ __forceinline__ v16bf b_from_f32(const float* __restrict__ p) {
  v16bf r;
#pragma unroll
  for (int j = 0; j < 16; ++j) r[j] = (__bf16)p[j];
  return r;
}

#define WMMA_BF16(A, B, C) \
  __builtin_amdgcn_wmma_f32_16x16x32_bf16(false, (A), false, (B), (short)0, (C), false, false)

// ---------- init: zero x0 slot + barrier counter ----------
__global__ void ssm_init_kernel(__bf16* __restrict__ xs, unsigned int* __restrict__ counter) {
  int i = blockIdx.x * blockDim.x + threadIdx.x;
  unsigned int* p = (unsigned int*)xs;
  for (int e = i; e < (BATCHSZ * SROWS) / 2; e += blockDim.x * gridDim.x) p[e] = 0u;
  if (i == 0) *counter = 0u;
}

// ---------- scan: x_{t+1} = M x_t + [B_re;B_im] u_t ----------
__global__ void CLUSTER_ATTR __launch_bounds__(128)
ssm_scan_kernel(const float* __restrict__ u,
                const float* __restrict__ A_re, const float* __restrict__ A_im,
                const float* __restrict__ B_re, const float* __restrict__ B_im,
                __bf16* __restrict__ xs, unsigned int* __restrict__ counter) {
  __shared__ alignas(16) __bf16 Msl[RPW * SROWS];     // 128 KB block-complex M slice
  __shared__ alignas(16) __bf16 Bsl[RPW * IN_F];      //  16 KB [B_re;B_im] slice
  __shared__ alignas(16) __bf16 Xsl[BATCHSZ * SROWS]; //  32 KB staged state X_t

  const int wg  = blockIdx.x;
  const int tid = threadIdx.x;
  const int R0  = wg * RPW;

  // Build bf16 weight slices in LDS once.
  for (int e = tid; e < RPW * SROWS; e += 128) {
    int rl = e >> 10, k = e & (SROWS - 1);
    int R  = R0 + rl;
    float v;
    if (R < NSTATE)
      v = (k < NSTATE) ?  A_re[(size_t)R * NSTATE + k]
                       : -A_im[(size_t)R * NSTATE + (k - NSTATE)];
    else {
      int Rp = R - NSTATE;
      v = (k < NSTATE) ?  A_im[(size_t)Rp * NSTATE + k]
                       :  A_re[(size_t)Rp * NSTATE + (k - NSTATE)];
    }
    Msl[e] = (__bf16)v;
  }
  for (int e = tid; e < RPW * IN_F; e += 128) {
    int rl = e >> 7, f = e & (IN_F - 1);
    int R  = R0 + rl;
    float v = (R < NSTATE) ? B_re[(size_t)R * IN_F + f]
                           : B_im[(size_t)(R - NSTATE) * IN_F + f];
    Bsl[e] = (__bf16)v;
  }
  __syncthreads();

  const int lane = tid & 31;
  const int wv   = tid >> 5;
  const int n    = lane & 15;           // output col == batch
  const int h    = lane >> 4;
  const int arow = wv * 16 + (lane & 15);

  for (int it = 0; it < T_LEN; ++it) {
    // ---- stage X_it (32 KB) into LDS: TDM if available, else cooperative copy
#if HAVE_TDM
    if (wv == 0) {
      unsigned long long ga = (unsigned long long)(uintptr_t)(xs + (size_t)it * BATCHSZ * SROWS);
      unsigned int lds_addr = (unsigned int)(uintptr_t)&Xsl[0];  // low 32 bits = LDS offset
      // D# group0: count=1, lds_addr, global_addr, type=2
      u32x4 g0 = { 1u, lds_addr,
                   (unsigned int)(ga & 0xFFFFFFFFull),
                   (unsigned int)((ga >> 32) & 0x01FFFFFFull) | 0x80000000u };
      // D# group1: data_size=4B; tensor_dim0=8192, tensor_dim1=1, tile_dim0=8192 (1-D tile),
      // tile_dim1/2=0 (unused), tensor_dim0_stride=8192
      i32x8 g1 = { 0x20000, 0x20000000, 0x10000, 0x20000000, 0, 8192, 0, 0 };
      // D# group2/3: degenerate higher dims (dim=1, tile=0, stride=0)
      i32x4 g2 = { 1, 1, 0, 0 };
      i32x4 g3 = { 0, 1 << 16, 0, 0 };
      i32x8 g4 = { 0, 0, 0, 0, 0, 0, 0, 0 };   // 6-arg toolchain variant: extra group
      __builtin_amdgcn_tensor_load_to_lds(g0, g1, g2, g3, g4, 0);
      __builtin_amdgcn_s_wait_tensorcnt(0);
    }
    __syncthreads();
#else
    {
      const uint4* src = (const uint4*)(xs + (size_t)it * BATCHSZ * SROWS);
      uint4* dst = (uint4*)Xsl;
      for (int e = tid; e < (BATCHSZ * SROWS) / 8; e += 128) dst[e] = src[e];
    }
    __syncthreads();
#endif

    const float* urow = u + ((size_t)n * T_LEN + it) * IN_F;

    v8f acc0 = {}, acc1 = {};
#pragma unroll 8
    for (int kt = 0; kt < 32; ++kt) {                 // K = 0..1023 (state)
      v16bf a = a_from_lds(&Msl[(size_t)arow * SROWS + kt * 32 + 16 * h]);
      v16bf b = *(const v16bf*)&Xsl[(size_t)n * SROWS + kt * 32 + 16 * h];
      if (kt & 1) acc1 = WMMA_BF16(a, b, acc1);
      else        acc0 = WMMA_BF16(a, b, acc0);
    }
#pragma unroll
    for (int kt = 0; kt < 4; ++kt) {                  // K = 1024..1151 (input)
      v16bf a = a_from_lds(&Bsl[(size_t)arow * IN_F + kt * 32 + 16 * h]);
      v16bf b = b_from_f32(urow + kt * 32 + 16 * h);
      if (kt & 1) acc1 = WMMA_BF16(a, b, acc1);
      else        acc0 = WMMA_BF16(a, b, acc0);
    }
    v8f acc = acc0 + acc1;

    // D-layout store: VGPR r -> row (8*h + r); 8 contiguous rows -> one 16B store.
    alignas(16) __bf16 tmp[8];
#pragma unroll
    for (int r = 0; r < 8; ++r) tmp[r] = (__bf16)acc[r];
    int mbase = R0 + wv * 16 + 8 * h;
    *(float4*)&xs[(size_t)((it + 1) * BATCHSZ + n) * SROWS + mbase] = *(const float4*)tmp;

    // ---- device-wide step barrier
    __threadfence();          // release: push stores to L2
    __syncthreads();
#if USE_CLUSTER
    if (wv == 0) {
      __builtin_amdgcn_s_cluster_barrier();           // signal -3 + wait -3
    } else {
      asm volatile("s_barrier_wait -3" ::: "memory");
    }
    __threadfence();          // acquire: invalidate WGP-level caches
#else
    if (tid == 0) {
      __hip_atomic_fetch_add(counter, 1u, __ATOMIC_RELEASE, __HIP_MEMORY_SCOPE_AGENT);
      unsigned int target = (unsigned int)(NWG * (it + 1));
      while (__hip_atomic_load(counter, __ATOMIC_ACQUIRE, __HIP_MEMORY_SCOPE_AGENT) < target)
        __builtin_amdgcn_s_sleep(1);
    }
    __syncthreads();
#endif
  }
}

// ---------- output: y = [C_re|-C_im] @ xs + u @ D^T + bias ----------
__global__ void __launch_bounds__(256)
ssm_out_kernel(const float* __restrict__ u,
               const float* __restrict__ C_re, const float* __restrict__ C_im,
               const float* __restrict__ D_w,  const float* __restrict__ bias,
               const __bf16* __restrict__ xs,  float* __restrict__ y) {
  const int ct   = blockIdx.x + 1;      // xs slot 1..2048 (y_t uses x_{t+1})
  const int t    = ct - 1;
  const int tid  = threadIdx.x;
  const int lane = tid & 31;
  const int ot   = tid >> 5;
  const int n    = lane & 15;
  const int h    = lane >> 4;

  const int o_a = ot * 16 + (lane & 15);
  const __bf16* xcol = xs + (size_t)(ct * BATCHSZ + n) * SROWS;
  const float*  urow = u + ((size_t)n * T_LEN + t) * IN_F;

  v8f acc0 = {}, acc1 = {};
#pragma unroll 8
  for (int kt = 0; kt < 32; ++kt) {
    v16bf a = a_from_cmat(C_re, C_im, o_a, kt * 32 + 16 * h);
    v16bf b = *(const v16bf*)&xcol[kt * 32 + 16 * h];
    if (kt & 1) acc1 = WMMA_BF16(a, b, acc1);
    else        acc0 = WMMA_BF16(a, b, acc0);
  }
#pragma unroll
  for (int kt = 0; kt < 4; ++kt) {
    v16bf a = a_from_f32(D_w + (size_t)o_a * IN_F, kt * 32 + 16 * h);
    v16bf b = b_from_f32(urow + kt * 32 + 16 * h);
    if (kt & 1) acc1 = WMMA_BF16(a, b, acc1);
    else        acc0 = WMMA_BF16(a, b, acc0);
  }
  v8f acc = acc0 + acc1;

  int obase = ot * 16 + 8 * h;
  alignas(32) float out8[8];
#pragma unroll
  for (int r = 0; r < 8; ++r) out8[r] = acc[r] + bias[obase + r];
  float* dst = y + ((size_t)n * T_LEN + t) * OUT_F + obase;
  *(float4*)(dst)     = *(const float4*)(out8);
  *(float4*)(dst + 4) = *(const float4*)(out8 + 4);
}

// ---------- host launcher ----------
extern "C" void kernel_launch(void* const* d_in, const int* in_sizes, int n_in,
                              void* d_out, int out_size, void* d_ws, size_t ws_size,
                              hipStream_t stream) {
  const float* u    = (const float*)d_in[0];
  const float* A_re = (const float*)d_in[1];
  const float* A_im = (const float*)d_in[2];
  const float* B_re = (const float*)d_in[3];
  const float* B_im = (const float*)d_in[4];
  const float* C_re = (const float*)d_in[5];
  const float* C_im = (const float*)d_in[6];
  const float* D_w  = (const float*)d_in[7];
  const float* bias = (const float*)d_in[8];
  float* y = (float*)d_out;

  __bf16* xs = (__bf16*)d_ws;
  size_t xs_bytes = (size_t)(T_LEN + 1) * BATCHSZ * SROWS * sizeof(__bf16);
  unsigned int* counter = (unsigned int*)((char*)d_ws + xs_bytes);

  ssm_init_kernel<<<dim3(64), dim3(256), 0, stream>>>(xs, counter);
  ssm_scan_kernel<<<dim3(NWG), dim3(128), 0, stream>>>(u, A_re, A_im, B_re, B_im, xs, counter);
  ssm_out_kernel<<<dim3(T_LEN), dim3(256), 0, stream>>>(u, C_re, C_im, D_w, bias, xs, y);
}
